// AttentionLayer_76605036692240
// MI455X (gfx1250) — compile-verified, matrix operand order checked by reading, not run on hardware
//
#include <hip/hip_runtime.h>
#include <hip/hip_bf16.h>

#define B_DIM 1024
#define S_DIM 512
#define D_DIM 200
#define A_DIM 100
#define KPAD  224   // D padded to multiple of 32
#define NPAD  112   // A padded to multiple of 16
#define KT    7     // KPAD/32
#define NT    7     // NPAD/16
#define ROWS  64    // S-rows per block (one 16-row M-tile per wave, 4 waves)
#define NCHUNK (S_DIM / 64)     // 8 chunks per batch
#define EPSV  1e-7f

typedef __attribute__((ext_vector_type(16))) __bf16 v16bf;
typedef __attribute__((ext_vector_type(2)))  __bf16 bf16x2;
typedef __attribute__((ext_vector_type(8)))  float  v8f;

__device__ __forceinline__ __bf16 f32_to_bf16_rne(float f) {
    union { float f; unsigned u; } v; v.f = f;
    unsigned r = v.u + 0x7FFFu + ((v.u >> 16) & 1u);   // round-to-nearest-even
    unsigned short h = (unsigned short)(r >> 16);
    __bf16 o; __builtin_memcpy(&o, &h, 2);
    return o;
}

// Packed f32x2 -> bf16x2. Prefer the dedicated builtin; otherwise use the
// C-level fptrunc cast: on gfx1250 ISel selects the hardware bf16 converter
// (v_cvt_pk_bf16_f32) even when the gfx950-gated builtin name is absent.
__device__ __forceinline__ bf16x2 cvt_pk_bf16(float a, float b) {
#if __has_builtin(__builtin_amdgcn_cvt_pk_bf16_f32)
    return __builtin_amdgcn_cvt_pk_bf16_f32(a, b);
#else
    bf16x2 r = { (__bf16)a, (__bf16)b };
    return r;
#endif
}

// CDNA5 hardware tanh (v_tanh_f32 trans op) when the builtin exists.
__device__ __forceinline__ float fast_tanhf(float x) {
#if __has_builtin(__builtin_amdgcn_tanhf)
    return __builtin_amdgcn_tanhf(x);
#elif __has_builtin(__builtin_amdgcn_tanh_f32)
    return __builtin_amdgcn_tanh_f32(x);
#else
    return tanhf(x);
#endif
}

// ---------------------------------------------------------------------------
// Kernel 0: pack W into WMMA B-fragment layout (bf16), pad bias/u.
// B-frag layout per (kt,nt): 32 lanes x 16 bf16, contiguous per lane:
//   lane<16 : j=0..7 -> K=kt*32+j      ; j=8..15 -> K=kt*32+16+(j-8)
//   lane>=16: j=0..7 -> K=kt*32+8+j    ; j=8..15 -> K=kt*32+24+(j-8)
//   N = nt*16 + (lane&15)
// ---------------------------------------------------------------------------
__global__ __launch_bounds__(256) void prep_kernel(
    const float* __restrict__ W, const float* __restrict__ bias,
    const float* __restrict__ u, __bf16* __restrict__ WB,
    float* __restrict__ biasPad, float* __restrict__ uPad)
{
    int i = blockIdx.x * 256 + threadIdx.x;
    if (i < KT * NT * 32 * 16) {
        int j    = i & 15;
        int l    = (i >> 4) & 31;
        int tile = i >> 9;
        int nt   = tile % NT;
        int kt   = tile / NT;
        int n    = nt * 16 + (l & 15);
        int kb   = (l < 16) ? 0 : 8;
        int k    = kt * 32 + kb + ((j < 8) ? j : j + 8);
        float v  = (k < D_DIM && n < A_DIM) ? W[k * A_DIM + n] : 0.0f;
        WB[i] = f32_to_bf16_rne(v);
    }
    if (i < NPAD) {
        biasPad[i] = (i < A_DIM) ? bias[i] : 0.0f;
        uPad[i]    = (i < A_DIM) ? u[i]    : 0.0f;
    }
}

// ---------------------------------------------------------------------------
// Kernel 1 (fused): for one (batch, 64-row chunk):
//   e[s]    = exp( sum_a tanh((x@W)[s,a]+bias[a]) * u[a] )   via bf16 WMMA
//   pOut[d] = sum_{s in chunk} x[s][d] * e[s]   (f32 x from LDS, single pass)
//   pSum    = sum_{s in chunk} e[s]
// x is read from HBM exactly once across the whole pipeline.
// ---------------------------------------------------------------------------
__global__ __launch_bounds__(128) void fused_kernel(
    const float* __restrict__ x, const __bf16* __restrict__ WB,
    const float* __restrict__ biasPad, const float* __restrict__ uPad,
    float* __restrict__ pOut, float* __restrict__ pSum)
{
    __shared__ __align__(16) float xs[ROWS][KPAD];   // 56 KB, f32, zero K-pad
    __shared__ float eld[ROWS];
    __shared__ float blockSum;

    const int b     = blockIdx.x;
    const int chunk = blockIdx.y;       // 0..7
    const int tid   = threadIdx.x;      // 128 threads = 4 waves
    const int row0  = chunk * ROWS;

    // Stage 64 rows of x (f32) at float4 granularity, zero-pad K 200->224
    const float* xb = x + (size_t)b * S_DIM * D_DIM;
    const int NQ = KPAD / 4;            // 56 float4 per padded row
    for (int idx = tid; idx < ROWS * NQ; idx += 128) {
        int r = idx / NQ, q = idx - r * NQ;
        float4 v;
        if (q < D_DIM / 4)
            v = *(const float4*)(xb + (size_t)(row0 + r) * D_DIM + q * 4);
        else
            v = make_float4(0.f, 0.f, 0.f, 0.f);
        *(float4*)&xs[r][q * 4] = v;    // global_load_b128 + ds_store_b128
    }
    if (tid == 0) blockSum = 0.0f;
    __syncthreads();

    const int wave = tid >> 5;          // 0..3 -> M tile
    const int lane = tid & 31;
    const int half = lane >> 4;         // 0 or 1
    const int lid  = lane & 15;
    const int mrow = wave * 16;

    // Build all 7 A-fragments once (hoisted across the N loop):
    // lane holds row M=lid; lanes<16: K = kt*32 + {0..7, 16..23}; lanes>=16: +8
    // Packed conversion lands pairs directly in fragment VGPRs.
    union AU { v16bf v; bf16x2 p[8]; };
    AU Afrag[KT];
    #pragma unroll
    for (int kt = 0; kt < KT; ++kt) {
        const int arow = mrow + lid;
        const int k0   = kt * 32 + half * 8;
        const float* ap0 = &xs[arow][k0];        // 8 f32 (ds_load_b128 x2)
        const float* ap1 = &xs[arow][k0 + 16];   // 8 f32
        #pragma unroll
        for (int q = 0; q < 4; ++q) {
            Afrag[kt].p[q]     = cvt_pk_bf16(ap0[2 * q], ap0[2 * q + 1]);
            Afrag[kt].p[4 + q] = cvt_pk_bf16(ap1[2 * q], ap1[2 * q + 1]);
        }
    }

    float partial[8];
    #pragma unroll
    for (int r = 0; r < 8; ++r) partial[r] = 0.0f;

    for (int nt = 0; nt < NT; ++nt) {
        v8f acc = {};
        #pragma unroll
        for (int kt = 0; kt < KT; ++kt) {
            // B fragment: 16 contiguous bf16 per lane, 32B aligned ->
            // two global_load_b128 straight into fragment VGPRs (L0/L2 hot)
            const v16bf Bf = *(const v16bf*)(WB +
                (((size_t)kt * NT + nt) * 32 + lane) * 16);
            acc = __builtin_amdgcn_wmma_f32_16x16x32_bf16(
                false, Afrag[kt].v, false, Bf, (short)0, acc, false, false);
        }
        // C layout: VGPR r, lanes 0-15 -> M=r, N=lid; lanes 16-31 -> M=8+r
        const int n    = nt * 16 + lid;
        const float bn = biasPad[n];
        const float un = uPad[n];       // zero on padded columns
        #pragma unroll
        for (int r = 0; r < 8; ++r)
            partial[r] += fast_tanhf(acc[r] + bn) * un;   // v_tanh_f32
    }

    // reduce the 16 lanes of each half-wave (same row, different N)
    #pragma unroll
    for (int r = 0; r < 8; ++r) {
        float v = partial[r];
        v += __shfl_xor(v, 1, 32);
        v += __shfl_xor(v, 2, 32);
        v += __shfl_xor(v, 4, 32);
        v += __shfl_xor(v, 8, 32);
        partial[r] = v;
    }
    if (lid == 0) {
        float wsum = 0.0f;
        #pragma unroll
        for (int r = 0; r < 8; ++r) {
            float e = expf(partial[r]);
            eld[mrow + half * 8 + r] = e;
            wsum += e;
        }
        atomicAdd(&blockSum, wsum);     // ds_add_f32
    }
    __syncthreads();

    // Chunk-partial weighted pooling from the f32 LDS tile (no 2nd x read)
    float* po = pOut + ((size_t)b * NCHUNK + chunk) * D_DIM;
    for (int t = tid; t < D_DIM; t += 128) {
        float acc = 0.0f;
        #pragma unroll 4
        for (int r = 0; r < ROWS; ++r)
            acc += xs[r][t] * eld[r];
        po[t] = acc;
    }
    if (tid == 0) pSum[(size_t)b * NCHUNK + chunk] = blockSum;
}

// ---------------------------------------------------------------------------
// Kernel 2: out[b][d] = (sum_c pOut[b][c][d]) / (sum_c pSum[b][c] + EPS)
// ---------------------------------------------------------------------------
__global__ __launch_bounds__(256) void finalize_kernel(
    const float* __restrict__ pOut, const float* __restrict__ pSum,
    float* __restrict__ out)
{
    const int b = blockIdx.x, t = threadIdx.x;
    float tot = 0.0f;
    #pragma unroll
    for (int c = 0; c < NCHUNK; ++c) tot += pSum[(size_t)b * NCHUNK + c];
    const float inv = 1.0f / (tot + EPSV);
    if (t < D_DIM) {
        float a = 0.0f;
        #pragma unroll
        for (int c = 0; c < NCHUNK; ++c)
            a += pOut[((size_t)b * NCHUNK + c) * D_DIM + t];
        out[(size_t)b * D_DIM + t] = a * inv;
    }
}

extern "C" void kernel_launch(void* const* d_in, const int* in_sizes, int n_in,
                              void* d_out, int out_size, void* d_ws, size_t ws_size,
                              hipStream_t stream)
{
    const float* x    = (const float*)d_in[0];   // [B,S,D]
    const float* W    = (const float*)d_in[1];   // [D,A]
    const float* bias = (const float*)d_in[2];   // [A]
    const float* u    = (const float*)d_in[3];   // [A,1]
    float* out        = (float*)d_out;           // [B,D]

    const int B = in_sizes[0] / (S_DIM * D_DIM); // 1024

    // workspace carve-up (all 16B aligned)
    char* ws = (char*)d_ws;
    __bf16* WB     = (__bf16*)(ws + 0);                 // KT*NT*512 bf16 = 50176 B
    float* biasPad = (float*)(ws + 50176);              // 112 f
    float* uPad    = (float*)(ws + 50624);              // 112 f
    float* pSum    = (float*)(ws + 51072);              // B*8 f      = 32 KB
    float* pOut    = (float*)(ws + 51072 + 32768);      // B*8*200 f  = 6.25 MB

    // 0) pack W / pad vectors
    prep_kernel<<<(KT * NT * 512 + 255) / 256, 256, 0, stream>>>(
        W, bias, u, WB, biasPad, uPad);

    // 1) fused scores + unnormalized pooling (single pass over x)
    dim3 g1(B, NCHUNK);
    fused_kernel<<<g1, 128, 0, stream>>>(x, WB, biasPad, uPad, pOut, pSum);

    // 2) reduce chunk partials and normalize
    finalize_kernel<<<B, 256, 0, stream>>>(pOut, pSum, out);
}
